// GATv2_35794257445084
// MI455X (gfx1250) — matrix operand order repaired and checked by reading, not compiled, not run on hardware
//
#include <hip/hip_runtime.h>
#include <math.h>

#define NN 1024
#define DD 128
#define ZZ 256

typedef __attribute__((ext_vector_type(16))) __bf16 v16bf;
typedef __attribute__((ext_vector_type(8)))  __bf16 v8bf;
typedef __attribute__((ext_vector_type(8)))  float  v8f;

static __device__ __forceinline__ v8f wmma_bf16(v16bf a, v16bf b, v8f c) {
  // D = A(16x32 bf16) * B(32x16 bf16) + C(16x16 f32)
  return __builtin_amdgcn_wmma_f32_16x16x32_bf16(false, a, false, b, (short)0, c, false, false);
}

// A-fragment (16x32, bf16) from a row-major bf16 tile (contiguous k order, same as the B packing):
// lane L<16 holds row M=L, K = {0..15}; lane L>=16 holds row M=L-16, K = {16..31}
static __device__ __forceinline__ v16bf load_a_bf16(const __bf16* tile, int ld, int lane, int k0) {
  int m  = lane & 15;
  int kb = k0 + ((lane >> 4) << 4);
  const __bf16* p = tile + (size_t)m * ld + kb;
  v8bf lo = *(const v8bf*)p;
  v8bf hi = *(const v8bf*)(p + 8);
  v16bf a;
#pragma unroll
  for (int t = 0; t < 8; ++t) { a[t] = lo[t]; a[t + 8] = hi[t]; }
  return a;
}

// Same A-fragment, but source is fp32 row-major; convert to bf16 in-register.
static __device__ __forceinline__ v16bf load_a_f32(const float* tile, int ld, int lane, int k0) {
  int m  = lane & 15;
  int kb = k0 + ((lane >> 4) << 4);
  const float* p = tile + (size_t)m * ld + kb;
  float4 l0 = *(const float4*)(p);
  float4 l1 = *(const float4*)(p + 4);
  float4 h0 = *(const float4*)(p + 8);
  float4 h1 = *(const float4*)(p + 12);
  v16bf a;
  a[0]=(__bf16)l0.x;  a[1]=(__bf16)l0.y;  a[2]=(__bf16)l0.z;  a[3]=(__bf16)l0.w;
  a[4]=(__bf16)l1.x;  a[5]=(__bf16)l1.y;  a[6]=(__bf16)l1.z;  a[7]=(__bf16)l1.w;
  a[8]=(__bf16)h0.x;  a[9]=(__bf16)h0.y;  a[10]=(__bf16)h0.z; a[11]=(__bf16)h0.w;
  a[12]=(__bf16)h1.x; a[13]=(__bf16)h1.y; a[14]=(__bf16)h1.z; a[15]=(__bf16)h1.w;
  return a;
}

// ---------- prep: pack z (bf16 row-major) and weights into B-fragment order ----------
// B-frag order: frag f, lane L, t -> B[k = f.k0 + ((L&16)?16:0) + t][n = f.n0 + (L&15)]
__global__ void k_prep_pack(const float* __restrict__ node, const float* __restrict__ hidden,
                            const float* __restrict__ Wm, const float* __restrict__ Wsk,
                            const float* __restrict__ W1, const float* __restrict__ W2,
                            const float* __restrict__ We,
                            __bf16* __restrict__ zb, __bf16* __restrict__ wallf,
                            __bf16* __restrict__ wef) {
  int gid = blockIdx.x * 256 + threadIdx.x;
  if (gid < 262144) {                             // z = [node | hidden], (1024 x 256) bf16
    int i = gid >> 8, d = gid & 255;
    float v = (d < 128) ? node[i * 128 + d] : hidden[i * 128 + (d - 128)];
    zb[gid] = (__bf16)v;
  } else if (gid < 262144 + 131072) {             // Wall = [Wm|Wskip|W1|W2] (256 x 512), frag order
    int e = gid - 262144;
    int t = e & 15, L = (e >> 4) & 31, ks = (e >> 9) & 7, nt = e >> 12;   // nt 0..31
    int c = nt * 16 + (L & 15);
    int k = ks * 32 + ((L & 16) ? 16 : 0) + t;
    const float* W = (c < 128) ? Wm : (c < 256) ? Wsk : (c < 384) ? W1 : W2;
    wallf[e] = (__bf16)W[k * 128 + (c & 127)];
  } else if (gid < 262144 + 131072 + 16384) {     // We (128 x 128), frag order
    int e = gid - (262144 + 131072);
    int t = e & 15, L = (e >> 4) & 31, ks = (e >> 9) & 3, n = e >> 11;    // n 0..7
    int mc = n * 16 + (L & 15);
    int k = ks * 32 + ((L & 16) ? 16 : 0) + t;
    wef[e] = (__bf16)We[k * 128 + mc];
  }
}

// pre_g[m] = graph @ Wg + bg + be   (be folded in here)
__global__ void k_preg(const float* __restrict__ g, const float* __restrict__ Wg,
                       const float* __restrict__ bg, const float* __restrict__ be,
                       float* __restrict__ preg) {
  int m = threadIdx.x;
  float s = bg[m] + be[m];
  for (int d = 0; d < 128; ++d) s += g[d] * Wg[d * 128 + m];
  preg[m] = s;
}

// ---------- fused z @ [Wm|Wskip|W1|W2] : (1024 x 256) x (256 x 512), bf16 WMMA ----------
__global__ void __launch_bounds__(256) k_gemm_zw(
    const __bf16* __restrict__ zb, const __bf16* __restrict__ wallf,
    const float* __restrict__ bm, const float* __restrict__ bsk,
    const float* __restrict__ b1, const float* __restrict__ b2,
    const float* __restrict__ preg,
    __bf16* __restrict__ vfrag, float* __restrict__ skip,
    float* __restrict__ pre1, float* __restrict__ pre2) {
  int lane = threadIdx.x & 31, wave = threadIdx.x >> 5;
  int job = blockIdx.x * 8 + wave;             // 2048 jobs = 64 row-tiles x 32 col-tiles
  int mt = job >> 5, nt = job & 31;
  const __bf16* ztile = zb + (size_t)mt * 16 * ZZ;
  v8f acc = {};
#pragma unroll
  for (int ks = 0; ks < 8; ++ks) {
    v16bf a = load_a_bf16(ztile, ZZ, lane, ks * 32);
    v16bf b = *(const v16bf*)(wallf + (((size_t)(nt * 8 + ks) * 32 + lane) << 4));
    acc = wmma_bf16(a, b, acc);
  }
  int c = nt * 16 + (lane & 15);               // global output column (0..511), region uniform per wave
  int reg = c >> 7, cc = c & 127;
#pragma unroll
  for (int r = 0; r < 8; ++r) {
    int i = mt * 16 + r + ((lane >> 4) << 3);  // D layout: lanes 0-15 M=r, lanes 16-31 M=r+8
    float xv = acc[r];
    if (reg == 0) {                            // values -> bf16, stored directly in B-frag order for P@V
      xv += bm[cc];
      int h = cc >> 4, hs = cc & 15;
      int idx = (((h * 32 + (i >> 5)) * 32 + (i & 16) + hs) << 4) + (i & 15);
      vfrag[idx] = (__bf16)xv;
    } else if (reg == 1) {
      skip[(size_t)i * 128 + cc] = xv + bsk[cc];
    } else if (reg == 2) {
      pre1[(size_t)i * 128 + cc] = xv + b1[cc] + preg[cc];   // pre_g + be folded in
    } else {
      pre2[(size_t)i * 128 + cc] = xv + b2[cc];
    }
  }
}

// ---------- the big one: pre_e GEMM + leaky-relu + per-head reduce -> logits ----------
__global__ void __launch_bounds__(256) k_logits(
    const float* __restrict__ edge, const int* __restrict__ adj,
    const __bf16* __restrict__ wef, const float* __restrict__ pre1,
    const float* __restrict__ pre2, const float* __restrict__ Avec,
    const float* __restrict__ Ab, float* __restrict__ logits) {
  __shared__ __bf16 wef_s[8 * 4 * 32 * 16];    // 32 KB: We in B-frag order
  __shared__ float combo[16 * 128];            // pre1 rows j0..j0+15 (pre_g/be folded)
  __shared__ float p2s[16 * 128];              // pre2 rows i0..i0+15
  int j0 = blockIdx.x * 16, i0 = blockIdx.y * 16;
  {
    const uint4* s = (const uint4*)wef; uint4* d = (uint4*)wef_s;
    for (int t = threadIdx.x; t < 2048; t += 256) d[t] = s[t];
    const float4* s1 = (const float4*)(pre1 + (size_t)j0 * 128); float4* d1 = (float4*)combo;
    const float4* s2 = (const float4*)(pre2 + (size_t)i0 * 128); float4* d2 = (float4*)p2s;
    for (int t = threadIdx.x; t < 512; t += 256) { d1[t] = s1[t]; d2[t] = s2[t]; }
  }
  __syncthreads();
  int lane = threadIdx.x & 31, wave = threadIdx.x >> 5;
  for (int mt = wave; mt < 16; mt += 8) {
    int i = i0 + mt;
    const float* etile = edge + ((size_t)i * NN + j0) * DD;   // rows = j0+dj, ld = 128
    v16bf af[4];
#pragma unroll
    for (int ks = 0; ks < 4; ++ks) af[ks] = load_a_f32(etile, DD, lane, ks * 32);
#pragma unroll
    for (int n = 0; n < 8; ++n) {              // n-tile == head
      v8f acc = {};
#pragma unroll
      for (int ks = 0; ks < 4; ++ks) {
        v16bf b = *(const v16bf*)(wef_s + (((n * 4 + ks) * 32 + lane) << 4));
        acc = wmma_bf16(af[ks], b, acc);
      }
      int ch = n * 16 + (lane & 15);           // MID channel of this lane's column
      float p2v = p2s[mt * 128 + ch];
      float aW = Avec[ch];                     // A[h][mh], row-major
#pragma unroll
      for (int r = 0; r < 8; ++r) {
        int dj = r + ((lane >> 4) << 3);
        float xv = acc[r] + combo[dj * 128 + ch] + p2v;
        float lr = fmaxf(xv, 0.01f * xv);      // leaky_relu, slope 0.01
        float part = lr * aW;
        part += __shfl_xor(part, 1, 32);       // reduce over 16 MID lanes
        part += __shfl_xor(part, 2, 32);
        part += __shfl_xor(part, 4, 32);
        part += __shfl_xor(part, 8, 32);
        if ((lane & 15) == 0) {
          int j = j0 + dj;
          float bias = (float)(adj[(size_t)i * NN + j] - 1) * 1e9f;
          logits[((size_t)n * NN + i) * NN + j] = part + Ab[n] + bias;
        }
      }
    }
  }
}

// ---------- row softmax (fp32) -> bf16 probs ----------
__global__ void __launch_bounds__(256) k_softmax(const float* __restrict__ logits,
                                                 __bf16* __restrict__ probs) {
  int row = blockIdx.x;                        // h*N + i
  const float* lp = logits + (size_t)row * NN;
  __bf16* pp = probs + (size_t)row * NN;
  int tid = threadIdx.x, lane = tid & 31, wave = tid >> 5;
  __shared__ float redm[8], reds[8];
  float x[4];
  float m = -3.4e38f;
#pragma unroll
  for (int k = 0; k < 4; ++k) { x[k] = lp[tid + k * 256]; m = fmaxf(m, x[k]); }
#pragma unroll
  for (int off = 16; off >= 1; off >>= 1) m = fmaxf(m, __shfl_xor(m, off, 32));
  if (lane == 0) redm[wave] = m;
  __syncthreads();
  m = redm[0];
#pragma unroll
  for (int w = 1; w < 8; ++w) m = fmaxf(m, redm[w]);
  float s = 0.f;
#pragma unroll
  for (int k = 0; k < 4; ++k) { x[k] = __expf(x[k] - m); s += x[k]; }
#pragma unroll
  for (int off = 16; off >= 1; off >>= 1) s += __shfl_xor(s, off, 32);
  if (lane == 0) reds[wave] = s;
  __syncthreads();
  s = 0.f;
#pragma unroll
  for (int w = 0; w < 8; ++w) s += reds[w];
  float inv = 1.f / s;
#pragma unroll
  for (int k = 0; k < 4; ++k) pp[tid + k * 256] = (__bf16)(x[k] * inv);
}

// ---------- ret = P @ V (per head, 16x1024 @ 1024x16 via 32 WMMAs) + skip, relu ----------
__global__ void __launch_bounds__(256) k_av(
    const __bf16* __restrict__ probs, const __bf16* __restrict__ vfrag,
    const float* __restrict__ skip, float* __restrict__ out) {
  int lane = threadIdx.x & 31, wave = threadIdx.x >> 5;
  int job = blockIdx.x * 8 + wave;             // 512 jobs = 8 heads x 64 row-tiles
  int h = job >> 6, it = job & 63;
  int i0 = it * 16;
  const __bf16* ptile = probs + ((size_t)h * NN + i0) * NN;
  v8f acc = {};
#pragma unroll 4
  for (int ks = 0; ks < 32; ++ks) {
    v16bf a = load_a_bf16(ptile, NN, lane, ks * 32);
    v16bf b = *(const v16bf*)(vfrag + (((size_t)(h * 32 + ks) * 32 + lane) << 4));
    acc = wmma_bf16(a, b, acc);
  }
  int hs = lane & 15, c = h * 16 + hs;
#pragma unroll
  for (int r = 0; r < 8; ++r) {
    int i = i0 + r + ((lane >> 4) << 3);
    float o = acc[r] + skip[(size_t)i * 128 + c];
    out[(size_t)i * 128 + c] = fmaxf(o, 0.f);
  }
}

extern "C" void kernel_launch(void* const* d_in, const int* in_sizes, int n_in,
                              void* d_out, int out_size, void* d_ws, size_t ws_size,
                              hipStream_t stream) {
  (void)in_sizes; (void)n_in; (void)out_size; (void)ws_size;
  const float* node   = (const float*)d_in[0];
  const float* edge   = (const float*)d_in[1];
  const float* graph  = (const float*)d_in[2];
  const int*   adj    = (const int*)d_in[3];
  const float* hidden = (const float*)d_in[4];
  const float* Wm  = (const float*)d_in[5];  const float* bm  = (const float*)d_in[6];
  const float* Wsk = (const float*)d_in[7];  const float* bsk = (const float*)d_in[8];
  const float* W1  = (const float*)d_in[9];  const float* b1  = (const float*)d_in[10];
  const float* W2  = (const float*)d_in[11]; const float* b2  = (const float*)d_in[12];
  const float* We  = (const float*)d_in[13]; const float* be  = (const float*)d_in[14];
  const float* Wg  = (const float*)d_in[15]; const float* bg  = (const float*)d_in[16];
  const float* Av  = (const float*)d_in[17]; const float* Ab  = (const float*)d_in[18];

  char* ws = (char*)d_ws;                       // ~50.6 MB used
  float*  logits = (float*) (ws + 0ull);        // 32 MB
  __bf16* probs  = (__bf16*)(ws + 33554432ull); // 16 MB
  __bf16* zb     = (__bf16*)(ws + 50331648ull); // 512 KB
  __bf16* wallf  = (__bf16*)(ws + 50855936ull); // 256 KB
  __bf16* wef    = (__bf16*)(ws + 51118080ull); // 32 KB
  __bf16* vfrag  = (__bf16*)(ws + 51150848ull); // 256 KB
  float*  pre1   = (float*) (ws + 51412992ull); // 512 KB
  float*  pre2   = (float*) (ws + 51937280ull); // 512 KB
  float*  skip   = (float*) (ws + 52461568ull); // 512 KB
  float*  preg   = (float*) (ws + 52985856ull); // 512 B
  float*  out    = (float*)d_out;

  k_prep_pack<<<1600, 256, 0, stream>>>(node, hidden, Wm, Wsk, W1, W2, We, zb, wallf, wef);
  k_preg<<<1, 128, 0, stream>>>(graph, Wg, bg, be, preg);
  k_gemm_zw<<<256, 256, 0, stream>>>(zb, wallf, bm, bsk, b1, b2, preg, vfrag, skip, pre1, pre2);
  k_logits<<<dim3(64, 64), 256, 0, stream>>>(edge, adj, wef, pre1, pre2, Av, Ab, logits);
  k_softmax<<<8192, 256, 0, stream>>>(logits, probs);
  k_av<<<64, 256, 0, stream>>>(probs, vfrag, skip, out);
}
